// Network_13649406066934
// MI455X (gfx1250) — matrix operand order repaired and checked by reading, not compiled, not run on hardware
//
#include <hip/hip_runtime.h>
#include <hip/hip_bf16.h>

typedef __attribute__((ext_vector_type(16))) _Float16 v16h;
typedef __attribute__((ext_vector_type(8)))  float    v8f;
typedef unsigned int u32x4 __attribute__((ext_vector_type(4)));
typedef int          i32x4 __attribute__((ext_vector_type(4)));
typedef int          i32x8 __attribute__((ext_vector_type(8)));

#define T_LEN 8192
#define F_LEN 26
#define B_LEN 64
#define CHUNK 512
#define WARM_TILES 4              // 64 warm-up steps: leak 0.4..0.8/step + ~per-step resets
#define NCHUNK (T_LEN / CHUNK)    // 16

// ---------------------------------------------------------------------------
// Setup: fold conv(1x1) + 3 linear layers into Weff[2][32] (f16, zero padded)
// and Keff[2] (f32).  out = h @ Weff^T + Keff, h_f = sum_c z_cf * conv_w_c.
// ---------------------------------------------------------------------------
__global__ void fold_weights_kernel(
    const float* __restrict__ conv_b,
    const float* __restrict__ w1, const float* __restrict__ b1,
    const float* __restrict__ w2, const float* __restrict__ b2,
    const float* __restrict__ w3, const float* __restrict__ b3,
    _Float16* __restrict__ WeffH, float* __restrict__ Keff)
{
    const int o = threadIdx.x;
    if (o >= 2) return;
    float w32[12];
    for (int k = 0; k < 12; ++k) {
        float s = 0.f;
        for (int j = 0; j < 4; ++j) s += w3[o * 4 + j] * w2[j * 12 + k];
        w32[k] = s;
    }
    float sumW = 0.f;
    for (int f = 0; f < 32; ++f) {
        float wf = 0.f;
        if (f < F_LEN)
            for (int k = 0; k < 12; ++k) wf += w32[k] * w1[k * F_LEN + f];
        WeffH[o * 32 + f] = (_Float16)wf;
        sumW += wf;
    }
    float kv = b3[o];
    for (int j = 0; j < 4; ++j)  kv += w3[o * 4 + j] * b2[j];
    for (int k = 0; k < 12; ++k) kv += w32[k] * b1[k];
    kv += conv_b[0] * sumW;      // conv bias folded through the linear chain
    Keff[o] = kv;
}

// ---------------------------------------------------------------------------
// TDM: async-load one 16(t) x 26(f) float tile of X into LDS.
// D# per CDNA5 ISA ch.8: group0 = {count, lds_addr, global_addr(57b), type=2},
// group1 = {data_size=4B, tensor_dim0=16, tensor_dim1=26, tile=16x26,
// dim0_stride=T}.  Groups 2/3 zero (2D tensor).  Tracked by TENSORcnt.
// ---------------------------------------------------------------------------
__device__ __forceinline__ void tdm_load_tile(unsigned lds_off,
                                              unsigned long long ga)
{
    u32x4 g0;
    g0[0] = 1u;                                   // count=1, user desc
    g0[1] = lds_off;                              // lds_addr (bytes)
    g0[2] = (unsigned)ga;                         // global_addr[31:0]
    g0[3] = ((unsigned)(ga >> 32) & 0x01FFFFFFu)  // global_addr[56:32]
            | (2u << 30);                         // type = 2 ("image")
    i32x8 g1;
    g1[0] = (int)(2u << 16);                      // data_size = 2 -> 4 bytes
    g1[1] = (int)(16u << 16);                     // tensor_dim0[15:0] = 16
    g1[2] = (int)((unsigned)F_LEN << 16);         // tensor_dim1[15:0] = 26
    g1[3] = (int)(16u << 16);                     // tile_dim0 = 16
    g1[4] = F_LEN;                                // tile_dim1 = 26, tile_dim2 = 0
    g1[5] = T_LEN;                                // tensor_dim0_stride[31:0] = 8192
    g1[6] = 0;                                    // stride0 hi / stride1 lo
    g1[7] = 0;
    i32x4 z4 = {0, 0, 0, 0};
#if __clang_major__ >= 23
    i32x8 z8 = {0, 0, 0, 0, 0, 0, 0, 0};
    __builtin_amdgcn_tensor_load_to_lds(g0, g1, z4, z4, z8, 0);
#else
    __builtin_amdgcn_tensor_load_to_lds(g0, g1, z4, z4, 0);
#endif
}

// ---------------------------------------------------------------------------
// Fused LIF scan + WMMA projection.  grid = (NCHUNK, B), block = 1 wave32.
// lane = feature f.  X tiles arrive via double-buffered TDM loads into LDS;
// every 16 steps the conv-weighted spike tile h[16t x 32f] is staged in LDS
// and projected with v_wmma_f32_16x16x32_f16 against Weff^T.
// ---------------------------------------------------------------------------
__global__ void __launch_bounds__(32)
lif_wmma_kernel(const float* __restrict__ X,      // [B,26,T]
                const float* __restrict__ tau,    // [3]
                const float* __restrict__ vth,    // [3]
                const float* __restrict__ cw,     // conv_w [3]
                const _Float16* __restrict__ WeffH,  // [2][32] padded
                const float* __restrict__ Keff,      // [2]
                float* __restrict__ out)          // [B,2,T]
{
    __shared__ __attribute__((aligned(16))) float xtile[2][F_LEN * 16]; // 2x1664B
    __shared__ __attribute__((aligned(16))) _Float16 hlds[16 * 32];     // 1024B

    const int L  = (int)threadIdx.x;      // lane 0..31
    const int b  = (int)blockIdx.y;
    const int ci = (int)blockIdx.x;
    const int f  = L;
    const bool fv = (f < F_LEN);

    const float dt0 = 0.001f * tau[0], dt1 = 0.001f * tau[1], dt2 = 0.001f * tau[2];
    const float th0 = vth[0], th1 = vth[1], th2 = vth[2];
    const float cw0 = cw[0],  cw1 = cw[1],  cw2 = cw[2];

    // --- B fragment: 16-bit B 32x16; lane N = L&15, K half picked by lane>=16
    const int N   = L & 15;
    const int hiK = (L >> 4) * 16;
    v16h bf;
#pragma unroll
    for (int j = 0; j < 16; ++j) {
        _Float16 wv = (_Float16)0.0f;
        if (N < 2) wv = WeffH[N * 32 + hiK + j];
        bf[j] = wv;
    }
    const float kN = (N < 2) ? Keff[N] : 0.f;

    float v0 = 0.f, v1 = 0.f, v2 = 0.f;
    const int tStart = ci * CHUNK;
    const int tEnd   = tStart + CHUNK;
    const int tWarm  = (ci == 0) ? tStart : (tStart - WARM_TILES * 16);

    const unsigned long long xbase =
        (unsigned long long)(uintptr_t)(X + (size_t)b * F_LEN * T_LEN);
    const unsigned lds0 = (unsigned)(uintptr_t)(&xtile[0][0]);
    const unsigned lds1 = (unsigned)(uintptr_t)(&xtile[1][0]);

    // Pre-issue first tile; double-buffer thereafter.
    tdm_load_tile(lds0, xbase + (unsigned long long)tWarm * 4ull);
    int nbuf = 0;

    for (int tt = tWarm; tt < tEnd; tt += 16) {
        const bool has_next = (tt + 16 < tEnd);     // uniform
        if (has_next)
            tdm_load_tile(nbuf ? lds0 : lds1,
                          xbase + (unsigned long long)(tt + 16) * 4ull);
        if (has_next) __builtin_amdgcn_s_wait_tensorcnt((short)1);
        else          __builtin_amdgcn_s_wait_tensorcnt((short)0);
        __asm__ __volatile__("" ::: "memory");      // LDS now holds tile nbuf

        // --- read my 16 x values (64B contiguous in LDS, row f of the tile)
        float xv[16];
        if (fv) {
            const float4* p = (const float4*)&xtile[nbuf][f * 16];
            float4 q0 = p[0], q1 = p[1], q2 = p[2], q3 = p[3];
            xv[0]=q0.x;  xv[1]=q0.y;  xv[2]=q0.z;  xv[3]=q0.w;
            xv[4]=q1.x;  xv[5]=q1.y;  xv[6]=q1.z;  xv[7]=q1.w;
            xv[8]=q2.x;  xv[9]=q2.y;  xv[10]=q2.z; xv[11]=q2.w;
            xv[12]=q3.x; xv[13]=q3.y; xv[14]=q3.z; xv[15]=q3.w;
        } else {
#pragma unroll
            for (int j = 0; j < 16; ++j) xv[j] = 0.f;
        }

        // --- 16 sequential LIF steps; h = conv-weighted spikes
#pragma unroll
        for (int j = 0; j < 16; ++j) {
            const float x = xv[j];
            v0 += dt0 * (x - v0);
            v1 += dt1 * (x - v1);
            v2 += dt2 * (x - v2);
            float h = 0.f;
            if (v0 > th0) { h += cw0; v0 = 0.f; }
            if (v1 > th1) { h += cw1; v1 = 0.f; }
            if (v2 > th2) { h += cw2; v2 = 0.f; }
            hlds[j * 32 + L] = (_Float16)(fv ? h : 0.f);
        }

        if (tt >= tStart) {   // uniform: warm-up tiles are not emitted
            // --- A fragment: 16-bit A 16x32; lane M = L&15,
            //     K = 2*hi8 + j (contiguous, matches B)
            const int M   = L & 15;
            const int hi8 = (L >> 4) * 8;
            v16h a;
#pragma unroll
            for (int j = 0; j < 16; ++j) {
                const int k = 2 * hi8 + j;
                a[j] = hlds[M * 32 + k];
            }
            v8f c = {};
            v8f d = __builtin_amdgcn_wmma_f32_16x16x32_f16(
                false, a, false, bf, (short)0, c, false, false);

            // --- D layout: lane N = L&15, row M = r + 8*(L>=16)
            if (N < 2) {
                float* op = out + ((size_t)b * 2 + N) * T_LEN + tt + 8 * (L >> 4);
#pragma unroll
                for (int r = 0; r < 8; ++r) op[r] = d[r] + kN;
            }
        }
        nbuf ^= 1;
    }
}

extern "C" void kernel_launch(void* const* d_in, const int* in_sizes, int n_in,
                              void* d_out, int out_size, void* d_ws, size_t ws_size,
                              hipStream_t stream) {
    const float* X    = (const float*)d_in[0];   // inputs [64,26,8192]
    const float* tau  = (const float*)d_in[1];   // [3]
    const float* vth  = (const float*)d_in[2];   // [3]
    const float* cw   = (const float*)d_in[3];   // conv_w [3]
    const float* cb   = (const float*)d_in[4];   // conv_b [1]
    const float* w1   = (const float*)d_in[5];
    const float* b1   = (const float*)d_in[6];
    const float* w2   = (const float*)d_in[7];
    const float* b2   = (const float*)d_in[8];
    const float* w3   = (const float*)d_in[9];
    const float* b3   = (const float*)d_in[10];

    _Float16* WeffH = (_Float16*)d_ws;                    // 64 halves = 128 B
    float*    Keff  = (float*)((char*)d_ws + 128);        // 2 floats

    fold_weights_kernel<<<1, 32, 0, stream>>>(cb, w1, b1, w2, b2, w3, b3,
                                              WeffH, Keff);

    dim3 grid(NCHUNK, B_LEN);   // (16, 64) -> 1024 wave32 workers
    lif_wmma_kernel<<<grid, 32, 0, stream>>>(X, tau, vth, cw, WeffH, Keff,
                                             (float*)d_out);
}
